// GNN_63677185131308
// MI455X (gfx1250) — compile-verified, matrix operand order checked by reading, not run on hardware
//
#include <hip/hip_runtime.h>

#define NN 10000
#define EE 320000
#define DD 128
#define HH 256
#define LL 4

typedef __attribute__((ext_vector_type(16))) __bf16 v16bf;
typedef __attribute__((ext_vector_type(8)))  __bf16 v8bf;
typedef __attribute__((ext_vector_type(8)))  float  v8f;

__device__ __forceinline__ float silu_f(float x) { return x / (1.0f + __expf(-x)); }

// Assemble a 16x32 bf16 A/B fragment half from two contiguous 16-byte LDS chunks:
// per documented layout, a lane holds k = base+0..7 and k = base+16..23.
__device__ __forceinline__ v16bf frag_cat(const __bf16* p) {
  v8bf lo = *(const v8bf*)p;
  v8bf hi = *(const v8bf*)(p + 16);
  return __builtin_shufflevector(lo, hi, 0,1,2,3,4,5,6,7,8,9,10,11,12,13,14,15);
}

__device__ __forceinline__ v8f wmma_bf16(v16bf a, v16bf b, v8f c) {
  return __builtin_amdgcn_wmma_f32_16x16x32_bf16(false, a, false, b, (short)0, c, false, false);
}

// ---- pack a [K,N] fp32 row-major weight into per-lane WMMA-B bf16 fragments ----
// tile index = kt*(N/16)+nt ; layout: dst[tile*512 + lane*16 + i]
__global__ void pack_b_kernel(const float* __restrict__ src, __bf16* __restrict__ dst,
                              int K, int N) {
  int tid = blockIdx.x * blockDim.x + threadIdx.x;
  int ntiles = (K / 32) * (N / 16);
  if (tid >= ntiles * 32) return;
  int tile = tid >> 5, lane = tid & 31;
  int nt = tile % (N / 16), kt = tile / (N / 16);
  int n = nt * 16 + (lane & 15);
  int kbase = kt * 32 + ((lane >= 16) ? 8 : 0);
  __bf16* out = dst + (size_t)tile * 512 + lane * 16;
#pragma unroll
  for (int i = 0; i < 16; ++i) {
    int k = kbase + (i < 8 ? i : i + 8);
    out[i] = (__bf16)src[(size_t)k * N + n];
  }
}

__global__ void zero_kernel(float* __restrict__ p, size_t n) {
  size_t i = (size_t)blockIdx.x * blockDim.x + threadIdx.x;
  size_t stride = (size_t)gridDim.x * blockDim.x;
  for (; i < n; i += stride) p[i] = 0.0f;
}

// ---- h = nodes @ emb_w + emb_b  (K=128, N=256), output bf16 ----
__global__ __launch_bounds__(128) void embed_kernel(
    const float* __restrict__ nodes, const __bf16* __restrict__ pw,
    const float* __restrict__ bias, __bf16* __restrict__ hout) {
  __shared__ __align__(16) __bf16 sA[16][DD];
  const int t = threadIdx.x, lane = t & 31, wave = t >> 5;
  const int m0 = blockIdx.x * 16;
  {
    int node = t >> 3, sub = t & 7;
    const float* src = nodes + (size_t)(m0 + node) * DD + sub * 16;
#pragma unroll
    for (int i = 0; i < 16; ++i) sA[node][sub * 16 + i] = (__bf16)src[i];
  }
  __syncthreads();
  const int nidx = lane & 15, koff = (lane >> 4) * 8;
  const int nb = wave * 4;
  v8f acc[4];
#pragma unroll
  for (int nt = 0; nt < 4; ++nt) {
    float bv = bias[(nb + nt) * 16 + nidx];
#pragma unroll
    for (int r = 0; r < 8; ++r) acc[nt][r] = bv;
  }
#pragma unroll
  for (int kt = 0; kt < DD / 32; ++kt) {
    v16bf a = frag_cat(&sA[nidx][kt * 32 + koff]);
#pragma unroll
    for (int nt = 0; nt < 4; ++nt) {
      v16bf b = *(const v16bf*)(pw + (size_t)(kt * (HH / 16) + nb + nt) * 512 + lane * 16);
      acc[nt] = wmma_bf16(a, b, acc[nt]);
    }
  }
#pragma unroll
  for (int nt = 0; nt < 4; ++nt)
#pragma unroll
    for (int r = 0; r < 8; ++r)
      hout[(size_t)(m0 + koff + r) * HH + (nb + nt) * 16 + nidx] = (__bf16)acc[nt][r];
}

// ---- fused edge MLP: 32 edges/block, 2 M-tiles per wave (B-fragment reuse x2) ----
// gather endpoints, 513->256 SiLU, 256->256 SiLU, atomic scatter (+edge_feat store)
__global__ __launch_bounds__(128) void edge_kernel(
    const __bf16* __restrict__ hbf, const int* __restrict__ edges,
    const float* __restrict__ eattr,
    const __bf16* __restrict__ pw1, const float* __restrict__ w1last,
    const float* __restrict__ b1,
    const __bf16* __restrict__ pw2, const float* __restrict__ b2,
    float* __restrict__ agg, float* __restrict__ mout) {
  __shared__ __align__(16) __bf16 sA[32][2 * HH];
  __shared__ __align__(16) __bf16 sT[32][HH];
  __shared__ int sRow[32];
  __shared__ float sEA[32];
  const int t = threadIdx.x, lane = t & 31, wave = t >> 5;
  const int e0 = blockIdx.x * 32;
  {
    int e = t >> 2, sub = t & 3;  // 4 threads per edge, 64 bf16 per half each
    int row = edges[e0 + e];
    int col = edges[EE + e0 + e];
    if (sub == 0) { sRow[e] = row; sEA[e] = eattr[e0 + e]; }
    const uint4* sr = (const uint4*)(hbf + (size_t)row * HH) + sub * 8;
    const uint4* sc = (const uint4*)(hbf + (size_t)col * HH) + sub * 8;
    uint4* dr = (uint4*)(&sA[e][0]) + sub * 8;
    uint4* dc = (uint4*)(&sA[e][HH]) + sub * 8;
#pragma unroll
    for (int i = 0; i < 8; ++i) { dr[i] = sr[i]; dc[i] = sc[i]; }
  }
  __syncthreads();
  const int nidx = lane & 15, koff = (lane >> 4) * 8;
  const int nb = wave * 4;
  v8f acc[2][4];
#pragma unroll
  for (int nt = 0; nt < 4; ++nt) {  // fold bias + edge_attr rank-1 (k=512 row of w1)
    int n = (nb + nt) * 16 + nidx;
    float bv = b1[n], wl = w1last[n];
#pragma unroll
    for (int mt = 0; mt < 2; ++mt)
#pragma unroll
      for (int r = 0; r < 8; ++r) acc[mt][nt][r] = bv + sEA[mt * 16 + koff + r] * wl;
  }
  for (int kt = 0; kt < (2 * HH) / 32; ++kt) {
    v16bf a0 = frag_cat(&sA[nidx][kt * 32 + koff]);
    v16bf a1 = frag_cat(&sA[16 + nidx][kt * 32 + koff]);
    const __bf16* bbase = pw1 + (size_t)(kt * (HH / 16) + nb) * 512 + lane * 16;
    __builtin_prefetch(bbase + (HH / 16) * 512, 0, 0);
#pragma unroll
    for (int nt = 0; nt < 4; ++nt) {
      v16bf b = *(const v16bf*)(bbase + nt * 512);
      acc[0][nt] = wmma_bf16(a0, b, acc[0][nt]);
      acc[1][nt] = wmma_bf16(a1, b, acc[1][nt]);
    }
  }
#pragma unroll
  for (int mt = 0; mt < 2; ++mt)
#pragma unroll
    for (int nt = 0; nt < 4; ++nt)
#pragma unroll
      for (int r = 0; r < 8; ++r)
        sT[mt * 16 + koff + r][(nb + nt) * 16 + nidx] = (__bf16)silu_f(acc[mt][nt][r]);
  __syncthreads();
  v8f acc2[2][4];
#pragma unroll
  for (int nt = 0; nt < 4; ++nt) {
    float bv = b2[(nb + nt) * 16 + nidx];
#pragma unroll
    for (int mt = 0; mt < 2; ++mt)
#pragma unroll
      for (int r = 0; r < 8; ++r) acc2[mt][nt][r] = bv;
  }
  for (int kt = 0; kt < HH / 32; ++kt) {
    v16bf a0 = frag_cat(&sT[nidx][kt * 32 + koff]);
    v16bf a1 = frag_cat(&sT[16 + nidx][kt * 32 + koff]);
    const __bf16* bbase = pw2 + (size_t)(kt * (HH / 16) + nb) * 512 + lane * 16;
    __builtin_prefetch(bbase + (HH / 16) * 512, 0, 0);
#pragma unroll
    for (int nt = 0; nt < 4; ++nt) {
      v16bf b = *(const v16bf*)(bbase + nt * 512);
      acc2[0][nt] = wmma_bf16(a0, b, acc2[0][nt]);
      acc2[1][nt] = wmma_bf16(a1, b, acc2[1][nt]);
    }
  }
#pragma unroll
  for (int mt = 0; mt < 2; ++mt) {
#pragma unroll
    for (int nt = 0; nt < 4; ++nt) {
#pragma unroll
      for (int r = 0; r < 8; ++r) {
        float x = silu_f(acc2[mt][nt][r]);
        int m = mt * 16 + koff + r;
        int n = (nb + nt) * 16 + nidx;
        atomicAdd(agg + (size_t)sRow[m] * HH + n, x);
        if (mout) mout[(size_t)(e0 + m) * HH + n] = x;
      }
    }
  }
}

// ---- fused node MLP: concat(h, agg) 512->256 SiLU, 256->256 (no act), bf16 out ----
__global__ __launch_bounds__(128) void node_kernel(
    const __bf16* __restrict__ hin, const float* __restrict__ agg,
    const __bf16* __restrict__ pw1, const float* __restrict__ b1,
    const __bf16* __restrict__ pw2, const float* __restrict__ b2,
    __bf16* __restrict__ hout) {
  __shared__ __align__(16) __bf16 sA[16][2 * HH];
  __shared__ __align__(16) __bf16 sT[16][HH];
  const int t = threadIdx.x, lane = t & 31, wave = t >> 5;
  const int m0 = blockIdx.x * 16;
  {
    int node = t >> 3, sub = t & 7;
    const uint4* sr = (const uint4*)(hin + (size_t)(m0 + node) * HH) + sub * 4;
    uint4* dr = (uint4*)(&sA[node][0]) + sub * 4;
#pragma unroll
    for (int i = 0; i < 4; ++i) dr[i] = sr[i];
    const float* ag = agg + (size_t)(m0 + node) * HH + sub * 32;
#pragma unroll
    for (int i = 0; i < 32; ++i) sA[node][HH + sub * 32 + i] = (__bf16)ag[i];
  }
  __syncthreads();
  const int nidx = lane & 15, koff = (lane >> 4) * 8;
  const int nb = wave * 4;
  v8f acc[4];
#pragma unroll
  for (int nt = 0; nt < 4; ++nt) {
    float bv = b1[(nb + nt) * 16 + nidx];
#pragma unroll
    for (int r = 0; r < 8; ++r) acc[nt][r] = bv;
  }
  for (int kt = 0; kt < (2 * HH) / 32; ++kt) {
    v16bf a = frag_cat(&sA[nidx][kt * 32 + koff]);
    const __bf16* bbase = pw1 + (size_t)(kt * (HH / 16) + nb) * 512 + lane * 16;
    __builtin_prefetch(bbase + (HH / 16) * 512, 0, 0);
#pragma unroll
    for (int nt = 0; nt < 4; ++nt) {
      v16bf b = *(const v16bf*)(bbase + nt * 512);
      acc[nt] = wmma_bf16(a, b, acc[nt]);
    }
  }
#pragma unroll
  for (int nt = 0; nt < 4; ++nt)
#pragma unroll
    for (int r = 0; r < 8; ++r)
      sT[koff + r][(nb + nt) * 16 + nidx] = (__bf16)silu_f(acc[nt][r]);
  __syncthreads();
  v8f acc2[4];
#pragma unroll
  for (int nt = 0; nt < 4; ++nt) {
    float bv = b2[(nb + nt) * 16 + nidx];
#pragma unroll
    for (int r = 0; r < 8; ++r) acc2[nt][r] = bv;
  }
  for (int kt = 0; kt < HH / 32; ++kt) {
    v16bf a = frag_cat(&sT[nidx][kt * 32 + koff]);
    const __bf16* bbase = pw2 + (size_t)(kt * (HH / 16) + nb) * 512 + lane * 16;
#pragma unroll
    for (int nt = 0; nt < 4; ++nt) {
      v16bf b = *(const v16bf*)(bbase + nt * 512);
      acc2[nt] = wmma_bf16(a, b, acc2[nt]);
    }
  }
#pragma unroll
  for (int nt = 0; nt < 4; ++nt)
#pragma unroll
    for (int r = 0; r < 8; ++r)
      hout[(size_t)(m0 + koff + r) * HH + (nb + nt) * 16 + nidx] = (__bf16)acc2[nt][r];
}

// ---- decoder: silu(h @ dw1 + db1) @ dw2 + db2 -> hd [N,3] ----
__global__ __launch_bounds__(128) void decoder_kernel(
    const __bf16* __restrict__ hin, const __bf16* __restrict__ pw1,
    const float* __restrict__ b1, const float* __restrict__ w2,
    const float* __restrict__ b2, float* __restrict__ hd_out,
    float* __restrict__ hd_ws) {
  __shared__ __align__(16) __bf16 sA[16][HH];
  __shared__ float sT[16][HH];
  const int t = threadIdx.x, lane = t & 31, wave = t >> 5;
  const int m0 = blockIdx.x * 16;
  {
    int node = t >> 3, sub = t & 7;
    const uint4* sr = (const uint4*)(hin + (size_t)(m0 + node) * HH) + sub * 4;
    uint4* dr = (uint4*)(&sA[node][0]) + sub * 4;
#pragma unroll
    for (int i = 0; i < 4; ++i) dr[i] = sr[i];
  }
  __syncthreads();
  const int nidx = lane & 15, koff = (lane >> 4) * 8;
  const int nb = wave * 4;
  v8f acc[4];
#pragma unroll
  for (int nt = 0; nt < 4; ++nt) {
    float bv = b1[(nb + nt) * 16 + nidx];
#pragma unroll
    for (int r = 0; r < 8; ++r) acc[nt][r] = bv;
  }
  for (int kt = 0; kt < HH / 32; ++kt) {
    v16bf a = frag_cat(&sA[nidx][kt * 32 + koff]);
    const __bf16* bbase = pw1 + (size_t)(kt * (HH / 16) + nb) * 512 + lane * 16;
#pragma unroll
    for (int nt = 0; nt < 4; ++nt) {
      v16bf b = *(const v16bf*)(bbase + nt * 512);
      acc[nt] = wmma_bf16(a, b, acc[nt]);
    }
  }
#pragma unroll
  for (int nt = 0; nt < 4; ++nt)
#pragma unroll
    for (int r = 0; r < 8; ++r)
      sT[koff + r][(nb + nt) * 16 + nidx] = silu_f(acc[nt][r]);
  __syncthreads();
  if (t < 48) {
    int r = t / 3, c = t % 3;
    float s = b2[c];
    for (int k = 0; k < HH; ++k) s += sT[r][k] * w2[k * 3 + c];
    hd_out[(size_t)(m0 + r) * 3 + c] = s;
    hd_ws[(size_t)(m0 + r) * 3 + c] = s;
  }
}

// ---- pairwise Euclidean distance, 64x64 tiles ----
__global__ __launch_bounds__(256) void dist_kernel(const float* __restrict__ hd,
                                                   float* __restrict__ dist) {
  __shared__ float si[64][3], sj[64][3];
  const int t = threadIdx.x;
  const int i0 = blockIdx.y * 64, j0 = blockIdx.x * 64;
  if (t < 192) {
    int r = t / 3, c = t % 3;
    si[r][c] = (i0 + r < NN) ? hd[(size_t)(i0 + r) * 3 + c] : 0.0f;
    sj[r][c] = (j0 + r < NN) ? hd[(size_t)(j0 + r) * 3 + c] : 0.0f;
  }
  __syncthreads();
  for (int idx = t; idx < 64 * 64; idx += 256) {
    int ii = idx >> 6, jj = idx & 63;
    int gi = i0 + ii, gj = j0 + jj;
    if (gi < NN && gj < NN) {
      float dx = si[ii][0] - sj[jj][0];
      float dy = si[ii][1] - sj[jj][1];
      float dz = si[ii][2] - sj[jj][2];
      dist[(size_t)gi * NN + gj] = sqrtf(dx * dx + dy * dy + dz * dz);
    }
  }
}

extern "C" void kernel_launch(void* const* d_in, const int* in_sizes, int n_in,
                              void* d_out, int out_size, void* d_ws, size_t ws_size,
                              hipStream_t stream) {
  (void)in_sizes; (void)n_in; (void)out_size; (void)ws_size;
  const float* nodes = (const float*)d_in[0];
  const int*   edges = (const int*)d_in[1];
  const float* eattr = (const float*)d_in[2];
  const float* emb_w = (const float*)d_in[3];
  const float* emb_b = (const float*)d_in[4];
  const float* ew1   = (const float*)d_in[5];
  const float* eb1   = (const float*)d_in[6];
  const float* ew2   = (const float*)d_in[7];
  const float* eb2   = (const float*)d_in[8];
  const float* nw1   = (const float*)d_in[9];
  const float* nb1   = (const float*)d_in[10];
  const float* nw2   = (const float*)d_in[11];
  const float* nb2   = (const float*)d_in[12];
  const float* dw1   = (const float*)d_in[13];
  const float* db1   = (const float*)d_in[14];
  const float* dw2   = (const float*)d_in[15];
  const float* db2   = (const float*)d_in[16];

  char* ws = (char*)d_ws;
  size_t off = 0;
  auto take = [&](size_t bytes) -> void* {
    void* p = ws + off;
    off += (bytes + 255) & ~(size_t)255;
    return p;
  };
  float*  agg   = (float*)take((size_t)NN * HH * sizeof(float));
  __bf16* hA    = (__bf16*)take((size_t)NN * HH * 2);
  __bf16* hB    = (__bf16*)take((size_t)NN * HH * 2);
  float*  hdws  = (float*)take((size_t)NN * 3 * sizeof(float));
  __bf16* p_emb = (__bf16*)take((size_t)DD * HH * 2);
  __bf16* p_ew1 = (__bf16*)take((size_t)LL * 2 * HH * HH * 2);
  __bf16* p_ew2 = (__bf16*)take((size_t)LL * HH * HH * 2);
  __bf16* p_nw1 = (__bf16*)take((size_t)LL * 2 * HH * HH * 2);
  __bf16* p_nw2 = (__bf16*)take((size_t)LL * HH * HH * 2);
  __bf16* p_dw1 = (__bf16*)take((size_t)HH * HH * 2);

  auto packGrid = [](int K, int N) { return ((K / 32) * (N / 16) * 32 + 255) / 256; };
  pack_b_kernel<<<packGrid(DD, HH), 256, 0, stream>>>(emb_w, p_emb, DD, HH);
  for (int l = 0; l < LL; ++l) {
    pack_b_kernel<<<packGrid(2 * HH, HH), 256, 0, stream>>>(
        ew1 + (size_t)l * (2 * HH + 1) * HH, p_ew1 + (size_t)l * 2 * HH * HH, 2 * HH, HH);
    pack_b_kernel<<<packGrid(HH, HH), 256, 0, stream>>>(
        ew2 + (size_t)l * HH * HH, p_ew2 + (size_t)l * HH * HH, HH, HH);
    pack_b_kernel<<<packGrid(2 * HH, HH), 256, 0, stream>>>(
        nw1 + (size_t)l * 2 * HH * HH, p_nw1 + (size_t)l * 2 * HH * HH, 2 * HH, HH);
    pack_b_kernel<<<packGrid(HH, HH), 256, 0, stream>>>(
        nw2 + (size_t)l * HH * HH, p_nw2 + (size_t)l * HH * HH, HH, HH);
  }
  pack_b_kernel<<<packGrid(HH, HH), 256, 0, stream>>>(dw1, p_dw1, HH, HH);

  embed_kernel<<<NN / 16, 128, 0, stream>>>(nodes, p_emb, emb_b, hA);

  __bf16* hcur = hA;
  __bf16* hnext = hB;
  float* edge_feat = (float*)d_out;
  for (int l = 0; l < LL; ++l) {
    zero_kernel<<<2048, 256, 0, stream>>>(agg, (size_t)NN * HH);
    edge_kernel<<<EE / 32, 128, 0, stream>>>(
        hcur, edges, eattr,
        p_ew1 + (size_t)l * 2 * HH * HH,
        ew1 + (size_t)l * (2 * HH + 1) * HH + (size_t)2 * HH * HH,  // w1 row k=512
        eb1 + (size_t)l * HH,
        p_ew2 + (size_t)l * HH * HH, eb2 + (size_t)l * HH,
        agg, (l == LL - 1) ? edge_feat : nullptr);
    node_kernel<<<NN / 16, 128, 0, stream>>>(
        hcur, agg,
        p_nw1 + (size_t)l * 2 * HH * HH, nb1 + (size_t)l * HH,
        p_nw2 + (size_t)l * HH * HH, nb2 + (size_t)l * HH, hnext);
    __bf16* tmp = hcur; hcur = hnext; hnext = tmp;
  }

  float* hd_out = (float*)d_out + (size_t)EE * HH;
  decoder_kernel<<<NN / 16, 128, 0, stream>>>(hcur, p_dw1, db1, dw2, db2, hd_out, hdws);

  float* dist = hd_out + (size_t)NN * 3;
  dim3 dg((NN + 63) / 64, (NN + 63) / 64);
  dist_kernel<<<dg, 256, 0, stream>>>(hdws, dist);
}
